// SceneLocalization_73967926771940
// MI455X (gfx1250) — matrix operand order repaired and checked by reading, not compiled, not run on hardware
//
#include <hip/hip_runtime.h>

typedef __attribute__((ext_vector_type(16))) __bf16 v16bf;
typedef __attribute__((ext_vector_type(8)))  __bf16 v8bf;
typedef __attribute__((ext_vector_type(8)))  float  v8f;
typedef __attribute__((ext_vector_type(4)))  float  v4f;

static constexpr int BATCH = 8192;
static constexpr int FEAT  = 7680;
static constexpr int HID   = 512;

__device__ __forceinline__ v8bf cvt8(v4f a, v4f b) {
  v8bf r;
  r[0] = (__bf16)a[0]; r[1] = (__bf16)a[1]; r[2] = (__bf16)a[2]; r[3] = (__bf16)a[3];
  r[4] = (__bf16)b[0]; r[5] = (__bf16)b[1]; r[6] = (__bf16)b[2]; r[7] = (__bf16)b[3];
  return r;
}

// Build a 16x32 bf16 A-fragment (exact wave32 lane layout: lane<16 holds K 0-7/16-23,
// lane>=16 holds K 8-15/24-31; caller passes ap already offset by c0).
__device__ __forceinline__ v16bf load_afrag_f32(const float* ap) {
  v4f a0 = *(const v4f*)ap;
  v4f a1 = *(const v4f*)(ap + 4);
  v4f a2 = *(const v4f*)(ap + 16);
  v4f a3 = *(const v4f*)(ap + 20);
  __builtin_prefetch(ap + 128, 0, 0);
  v8bf lo = cvt8(a0, a1);
  v8bf hi = cvt8(a2, a3);
  return __builtin_shufflevector(lo, hi, 0,1,2,3,4,5,6,7,8,9,10,11,12,13,14,15);
}

__device__ __forceinline__ v16bf load_afrag_bf16(const __bf16* ap) {
  v8bf lo = *(const v8bf*)ap;
  v8bf hi = *(const v8bf*)(ap + 16);
  return __builtin_shufflevector(lo, hi, 0,1,2,3,4,5,6,7,8,9,10,11,12,13,14,15);
}

// D = relu(A @ W^T + bias). A: [*, KDIM] (f32 or bf16), W: [HID, KDIM] f32 (torch layout),
// out: [*, HID] bf16. Workgroup: 256 M x 64 N (8 waves, each 32 M x 64 N, 8 accumulators).
// W sub-tile staged fp32->bf16 into double-buffered LDS in exact B-fragment layout;
// one barrier per K-step; 8 back-to-back WMMAs per step.
template <bool A_IS_F32, int KDIM>
__global__ __launch_bounds__(256)
void gemm_relu_bf16_wmma(const void* __restrict__ Aptr, const float* __restrict__ W,
                         const float* __restrict__ bias, __bf16* __restrict__ out) {
  __shared__ __align__(32) __bf16 Bs[2][4][32][16];  // double buffer, 8 KB total

  const int lane = threadIdx.x & 31;
  const int wv   = threadIdx.x >> 5;
  const int M0   = blockIdx.y * 256;
  const int N0   = blockIdx.x * 64;
  const int rowA = M0 + wv * 32 + (lane & 15);

  // cooperative B staging: 256 threads fill 4 tiles x 32 lane-slots x 16 bf16
  const int bt  = threadIdx.x >> 6;                  // N-tile 0..3
  const int br  = threadIdx.x & 63;
  const int bl  = br >> 1;                           // fragment lane slot 0..31
  const int bh  = br & 1;                            // which 8-element half
  const int bn  = N0 + bt * 16 + (bl & 15);          // output column (W row)
  const int bk0 = ((bl < 16) ? 0 : 8) + bh * 16;     // K offset inside 32-wide step
  const float* wrow = W + (size_t)bn * KDIM;

  const int c0 = (lane < 16) ? 0 : 8;                // A-fragment K sub-offset

  v8f acc[2][4] = {};

  // prologue: stage K-step 0
  {
    const float* wp = wrow + bk0;
    v4f w0 = *(const v4f*)wp;
    v4f w1 = *(const v4f*)(wp + 4);
    *(v8bf*)&Bs[0][bt][bl][bh * 8] = cvt8(w0, w1);
  }
  __syncthreads();

  constexpr int NITER = KDIM / 32;
  for (int i = 0; i < NITER; ++i) {
    const int kb = i * 32;
    const int p  = i & 1;
    const bool has_next = (i + 1 < NITER);

    // issue next tile's global loads early (latency hidden under WMMAs)
    v4f nw0, nw1;
    if (has_next) {
      const float* wp = wrow + kb + 32 + bk0;
      nw0 = *(const v4f*)wp;
      nw1 = *(const v4f*)(wp + 4);
    }

    // all 4 B fragments into distinct registers -> waits hoist, WMMAs go back-to-back
    v16bf bfrag[4];
#pragma unroll
    for (int t = 0; t < 4; ++t) bfrag[t] = *(const v16bf*)&Bs[p][t][lane][0];

    v16bf afrag[2];
#pragma unroll
    for (int m = 0; m < 2; ++m) {
      if constexpr (A_IS_F32) {
        const float* ap = (const float*)Aptr + (size_t)(rowA + m * 16) * KDIM + kb + c0;
        afrag[m] = load_afrag_f32(ap);
      } else {
        const __bf16* ap = (const __bf16*)Aptr + (size_t)(rowA + m * 16) * KDIM + kb + c0;
        afrag[m] = load_afrag_bf16(ap);
      }
    }

#pragma unroll
    for (int m = 0; m < 2; ++m)
#pragma unroll
      for (int t = 0; t < 4; ++t)
        acc[m][t] = __builtin_amdgcn_wmma_f32_16x16x32_bf16(
            /*neg_a=*/false, afrag[m], /*neg_b=*/false, bfrag[t],
            /*c_mod=*/(short)0, acc[m][t], /*reuse_a=*/false, /*reuse_b=*/false);

    // stage next tile into the other buffer (its readers finished before last barrier)
    if (has_next) *(v8bf*)&Bs[p ^ 1][bt][bl][bh * 8] = cvt8(nw0, nw1);
    __syncthreads();
  }

  // epilogue: bias + ReLU + bf16 store (C/D layout: VGPR v -> M=v or v+8)
#pragma unroll
  for (int m = 0; m < 2; ++m) {
    const int mrow = M0 + wv * 32 + m * 16 + ((lane < 16) ? 0 : 8);
#pragma unroll
    for (int t = 0; t < 4; ++t) {
      const int n   = N0 + t * 16 + (lane & 15);
      const float bv = bias[n];
#pragma unroll
      for (int v = 0; v < 8; ++v) {
        float x = acc[m][t][v] + bv;
        x = fmaxf(x, 0.0f);
        out[(size_t)(mrow + v) * HID + n] = (__bf16)x;
      }
    }
  }
}

// shared = h2 @ W3^T + b3 ([B,4]), then per-scene routed 2x2 affine + rotation.
// One wave32 per row; vector bf16/f32 loads + shuffle reduction.
__global__ __launch_bounds__(256)
void head_routed_kernel(const __bf16* __restrict__ h2, const float* __restrict__ W3,
                        const float* __restrict__ b3, const long long* __restrict__ sidx,
                        const float* __restrict__ xyW, const float* __restrict__ xyb,
                        const float* __restrict__ tW, float* __restrict__ out) {
  const int lane = threadIdx.x & 31;
  const int wv   = threadIdx.x >> 5;
  const int row  = blockIdx.x * 8 + wv;
  const __bf16* hrow = h2 + (size_t)row * HID;

  float p[4] = {0.f, 0.f, 0.f, 0.f};
#pragma unroll
  for (int c = 0; c < 2; ++c) {
    const int k = c * 256 + lane * 8;
    v8bf hv = *(const v8bf*)(hrow + k);
#pragma unroll
    for (int o = 0; o < 4; ++o) {
      v4f wlo = *(const v4f*)(W3 + o * HID + k);
      v4f whi = *(const v4f*)(W3 + o * HID + k + 4);
      float s = 0.f;
#pragma unroll
      for (int j = 0; j < 4; ++j)
        s += (float)hv[j] * wlo[j] + (float)hv[j + 4] * whi[j];
      p[o] += s;
    }
  }
#pragma unroll
  for (int off = 16; off > 0; off >>= 1)
#pragma unroll
    for (int o = 0; o < 4; ++o) p[o] += __shfl_down(p[o], off, 32);

  if (lane == 0) {
    const float s0 = p[0] + b3[0];
    const float s1 = p[1] + b3[1];
    const float s2 = p[2] + b3[2];
    const float s3 = p[3] + b3[3];
    const int si = (int)sidx[row];
    const float* wxy = xyW + (size_t)si * 4;
    const float* bx  = xyb + (size_t)si * 2;
    const float* wt  = tW  + (size_t)si * 4;
    v4f o;
    o[0] = wxy[0] * s0 + wxy[1] * s1 + bx[0];
    o[1] = wxy[2] * s0 + wxy[3] * s1 + bx[1];
    o[2] = wt[0] * s2 + wt[1] * s3;
    o[3] = wt[2] * s2 + wt[3] * s3;
    *(v4f*)(out + (size_t)row * 4) = o;
  }
}

extern "C" void kernel_launch(void* const* d_in, const int* in_sizes, int n_in,
                              void* d_out, int out_size, void* d_ws, size_t ws_size,
                              hipStream_t stream) {
  const float*     feat = (const float*)d_in[0];
  const long long* sidx = (const long long*)d_in[1];   // int64 in reference
  const float*     W1   = (const float*)d_in[2];
  const float*     b1   = (const float*)d_in[3];
  const float*     W2   = (const float*)d_in[4];
  const float*     b2   = (const float*)d_in[5];
  const float*     W3   = (const float*)d_in[6];
  const float*     b3   = (const float*)d_in[7];
  const float*     xyW  = (const float*)d_in[8];
  const float*     xyb  = (const float*)d_in[9];
  const float*     tW   = (const float*)d_in[10];
  float* out = (float*)d_out;

  __bf16* h1 = (__bf16*)d_ws;                           // [8192, 512] bf16 (8 MB)
  __bf16* h2 = h1 + (size_t)BATCH * HID;                // [8192, 512] bf16 (8 MB)

  dim3 grid1(HID / 64, BATCH / 256);                    // (8, 32)
  gemm_relu_bf16_wmma<true,  FEAT><<<grid1, 256, 0, stream>>>(feat, W1, b1, h1);
  gemm_relu_bf16_wmma<false, HID ><<<grid1, 256, 0, stream>>>(h1,   W2, b2, h2);
  head_routed_kernel<<<BATCH / 8, 256, 0, stream>>>(h2, W3, b3, sidx, xyW, xyb, tW, out);
}
